// MultichannelMultiheadAttention_12919261626435
// MI455X (gfx1250) — compile-verified, matrix operand order checked by reading, not run on hardware
//
#include <hip/hip_runtime.h>
#include <math.h>
#include <stdint.h>

#define DI __device__ __forceinline__

typedef __attribute__((ext_vector_type(16))) _Float16 v16h;
typedef __attribute__((ext_vector_type(8)))  _Float16 v8h;
typedef __attribute__((ext_vector_type(4)))  _Float16 v4h;
typedef __attribute__((ext_vector_type(2)))  _Float16 v2h;
typedef __attribute__((ext_vector_type(8)))  float    v8f;
typedef __attribute__((ext_vector_type(4)))  float    v4f;
typedef __attribute__((ext_vector_type(4)))  unsigned int v4u;
typedef __attribute__((ext_vector_type(4)))  int      v4i;
typedef __attribute__((ext_vector_type(8)))  int      v8i;

// Problem constants (from reference)
constexpr int BB  = 2;
constexpr int CC  = 4;
constexpr int FF  = 512;
constexpr int WW  = 1024;
constexpr int NHH = 8;
constexpr int DHH = 64;
constexpr int BCC = BB * CC;   // 8 batch*channel pairs

// ---- Tensor Data Mover availability -----------------------------------------
#if defined(__has_builtin)
#if __has_builtin(__builtin_amdgcn_tensor_load_to_lds) && \
    __has_builtin(__builtin_amdgcn_s_wait_tensorcnt)
#define USE_TDM 1
#endif
#endif
#ifndef USE_TDM
#define USE_TDM 0
#endif

DI v8f vzero8() {
  v8f z;
#pragma unroll
  for (int i = 0; i < 8; ++i) z[i] = 0.0f;
  return z;
}

// Load one 16x32 f16 WMMA A/B fragment from an LDS row-major tile.
// CDNA5 16-bit layout: lane L(0-15): K {0..7, 16..23}; lane L+16: K {8..15, 24..31}.
DI v16h load_frag16(const _Float16* rowptr, int lane_hi) {
  const _Float16* p = rowptr + lane_hi * 8;
  v8h lo = *(const v8h*)(p);
  v8h hi = *(const v8h*)(p + 16);
  v16h r;
#pragma unroll
  for (int i = 0; i < 8; ++i) { r[i] = lo[i]; r[i + 8] = hi[i]; }
  return r;
}

DI v8f wmma_f16(v16h a, v16h b, v8f c) {
  return __builtin_amdgcn_wmma_f32_16x16x32_f16(
      false, a, false, b, (short)0, c, false, false);
}

#if USE_TDM
// 2D f16 tile DMA: global (row-major, stride0 elems/row) -> LDS with HW row pad.
// padi/pada are the encoded D# pad_interval / pad_amount fields.
DI void tdm_load_2d_f16(void* lds_dst, const _Float16* gsrc,
                        unsigned tile0, unsigned tile1,
                        unsigned tdim0, unsigned tdim1,
                        unsigned stride0, unsigned padi, unsigned pada) {
  unsigned lds_off = (unsigned)(size_t)lds_dst;           // low 32 bits = LDS byte offset
  unsigned long long ga = (unsigned long long)(size_t)gsrc;
  v4u g0;
  g0[0] = 1u;                                             // count=1, user descriptor
  g0[1] = lds_off;                                        // lds_addr
  g0[2] = (unsigned)ga;                                   // global_addr[31:0]
  g0[3] = (unsigned)(ga >> 32) | (2u << 30);              // global_addr[56:32] | type=2
  v8i g1;
  g1[0] = (int)((1u << 16) |                              // data_size = 2 bytes
                (1u << 20) |                              // pad_enable
                (padi << 22) | (pada << 25));             // pad interval / amount
  g1[1] = (int)(tdim0 << 16);                             // tensor_dim0 (bits 79:48) low
  g1[2] = (int)((tdim0 >> 16) | (tdim1 << 16));           // dim0 hi | tensor_dim1 low
  g1[3] = (int)((tdim1 >> 16) | (tile0 << 16));           // dim1 hi | tile_dim0
  g1[4] = (int)(tile1);                                   // tile_dim1 (tile_dim2 = 0)
  g1[5] = (int)(stride0);                                 // tensor_dim0_stride[31:0]
  g1[6] = 0;                                              // stride0 hi | dim1_stride lo
  g1[7] = 0;
  v4i gz = {0, 0, 0, 0};
#if defined(__clang_major__) && __clang_major__ >= 23
  v8i gz8 = {0, 0, 0, 0, 0, 0, 0, 0};
  __builtin_amdgcn_tensor_load_to_lds(g0, g1, gz, gz, gz8, 0);
#else
  __builtin_amdgcn_tensor_load_to_lds(g0, g1, gz, gz, 0);
#endif
}
#endif

// ---------------------------------------------------------------------------
// Per-(b,c) GEMM: Out(512 x 1024) = Wc(512x512) @ X(512x1024) + bias
// Block tile 128x128, 8 waves (2 x 4), wave tile 64x32, K-step 32.
// TRANS_OUT: head-transposed (bc,h,w,d) layout for Q/K.  SCALE: fold 1/sqrt(F).
// TIN: float (x) or _Float16 (attention result). TOUT: _Float16 (q/k/v) or float.
// ---------------------------------------------------------------------------
template <bool TRANS_OUT, bool SCALE, typename TIN, typename TOUT>
__global__ __launch_bounds__(256) void mc_gemm_wmma(
    const float* __restrict__ Wt,    // (C, F, F)
    const float* __restrict__ Bias,  // (C, F)
    const TIN* __restrict__ X,       // (BC, F, W)
    TOUT* __restrict__ Out) {
  __shared__ _Float16 aS[128][40];   // [m][k], +8 pad
  __shared__ _Float16 bS[128][40];   // [n][k] (x transposed into LDS)

  const int bc   = blockIdx.z;
  const int c    = bc % CC;
  const int Mb   = blockIdx.y * 128;
  const int Nb   = blockIdx.x * 128;
  const int t    = threadIdx.x;
  const int lane = t & 31;
  const int wv   = t >> 5;
  const int wm   = wv & 1;
  const int wn   = wv >> 1;
  const int ln15 = lane & 15;
  const int lhi  = lane >> 4;

  const float* Wc = Wt + (size_t)c * FF * FF;
  const TIN*   Xb = X + (size_t)bc * FF * WW;

  v8f acc[4][2];
#pragma unroll
  for (int i = 0; i < 4; ++i)
#pragma unroll
    for (int j = 0; j < 2; ++j) acc[i][j] = vzero8();

  for (int kb = 0; kb < FF; kb += 32) {
    __syncthreads();
    // A tile: weights 128x32 fp32 -> f16
#pragma unroll
    for (int i = 0; i < 4; ++i) {
      int u   = t + i * 256;
      int row = u >> 3;
      int c4  = (u & 7) << 2;
      const float* gp = Wc + (size_t)(Mb + row) * FF + kb + c4;
      v4f d = *(const v4f*)gp;
      if (kb + 32 < FF) __builtin_prefetch(gp + 32, 0, 1);   // next K tile
      v4h h;
#pragma unroll
      for (int e = 0; e < 4; ++e) h[e] = (_Float16)d[e];
      *(v4h*)&aS[row][c4] = h;
    }
    // B tile: X[kb..kb+31][Nb..Nb+127] -> bS[n][k] (transposed in LDS)
    if constexpr (sizeof(TIN) == 4) {
#pragma unroll
      for (int i = 0; i < 4; ++i) {
        int u    = t + i * 256;
        int krow = u >> 5;
        int c4   = (u & 31) << 2;
        v4f d = *(const v4f*)(Xb + (size_t)(kb + krow) * WW + Nb + c4);
#pragma unroll
        for (int e = 0; e < 4; ++e) bS[c4 + e][krow] = (_Float16)d[e];
      }
    } else {
#pragma unroll
      for (int i = 0; i < 2; ++i) {
        int u    = t + i * 256;                // 512 v8h chunks cover 32x128
        int krow = u >> 4;
        int c8   = (u & 15) << 3;
        v8h d = *(const v8h*)(Xb + (size_t)(kb + krow) * WW + Nb + c8);
#pragma unroll
        for (int e = 0; e < 8; ++e) bS[c8 + e][krow] = d[e];
      }
    }
    __syncthreads();

    v16h bfr[2];
#pragma unroll
    for (int j = 0; j < 2; ++j)
      bfr[j] = load_frag16(&bS[wn * 32 + j * 16 + ln15][0], lhi);
#pragma unroll
    for (int i = 0; i < 4; ++i) {
      v16h afr = load_frag16(&aS[wm * 64 + i * 16 + ln15][0], lhi);
#pragma unroll
      for (int j = 0; j < 2; ++j) acc[i][j] = wmma_f16(afr, bfr[j], acc[i][j]);
    }
  }

  const float* bias = Bias + (size_t)c * FF;
  const float osc = SCALE ? 0.044194173824159216f : 1.0f;  // 1/sqrt(512)
#pragma unroll
  for (int i = 0; i < 4; ++i) {
    int mbase = Mb + wm * 64 + i * 16 + 8 * lhi;
#pragma unroll
    for (int j = 0; j < 2; ++j) {
      int n = Nb + wn * 32 + j * 16 + ln15;
#pragma unroll
      for (int r = 0; r < 8; ++r) {
        int m = mbase + r;
        float val = (acc[i][j][r] + bias[m]) * osc;
        if (TRANS_OUT) {
          int h = m >> 6, d = m & 63;
          Out[((size_t)(bc * NHH + h) * WW + n) * DHH + d] = (TOUT)val;
        } else {
          Out[(size_t)bc * FF * WW + (size_t)m * WW + n] = (TOUT)val;
        }
      }
    }
  }
}

// ---------------------------------------------------------------------------
// RoPE on head-transposed f16 Q/K: layout (bc, h, w, d), interleaved pairs.
// ---------------------------------------------------------------------------
__global__ __launch_bounds__(256) void rope_kernel(
    _Float16* __restrict__ q, _Float16* __restrict__ k,
    const float* __restrict__ freqs) {
  int idx = blockIdx.x * 256 + threadIdx.x;   // BC*NH*W*(DH/2) = 2^21 threads
  int p  = idx & 31;
  int w  = (idx >> 5) & (WW - 1);
  int hw = idx >> 15;
  if (hw >= BCC * NHH) return;
  float ang = (float)w * freqs[p];
  float s, cth;
  __sincosf(ang, &s, &cth);
  size_t base = ((size_t)hw * WW + w) * DHH + 2 * p;
  v2h qv = *(v2h*)(q + base);
  v2h kv = *(v2h*)(k + base);
  float q0 = (float)qv[0], q1 = (float)qv[1];
  float k0 = (float)kv[0], k1 = (float)kv[1];
  v2h qo, ko;
  qo[0] = (_Float16)(q0 * cth - q1 * s);
  qo[1] = (_Float16)(q1 * cth + q0 * s);
  ko[0] = (_Float16)(k0 * cth - k1 * s);
  ko[1] = (_Float16)(k1 * cth + k0 * s);
  *(v2h*)(q + base) = qo;
  *(v2h*)(k + base) = ko;
}

// ---------------------------------------------------------------------------
// Flash attention per (bc, head): 128-query tile, 128-key chunks.
// Q/K f16 in (bc,h,w,d) (Q pre-scaled); V f16 in (bc,f,w) = per-head V^T.
// K/V chunks ping-pong through double-buffered LDS via the Tensor Data Mover:
// chunk j+1's DMA is issued before chunk j's compute; s_wait_tensorcnt(2)
// proves chunk j landed (TDM ops from one wave complete in order) while
// chunk j+1 streams in behind the WMMA work.
// ---------------------------------------------------------------------------
__global__ __launch_bounds__(256) void attn_wmma(
    const _Float16* __restrict__ Q, const _Float16* __restrict__ K,
    const _Float16* __restrict__ V, _Float16* __restrict__ A) {
  __shared__ _Float16 qS[128][72];          // [q][d]     row = 128B + 16B pad
  __shared__ _Float16 kS[2][128][72];       // [key][d]   double-buffered
  __shared__ _Float16 vS[2][64][136];       // [d][key]   double-buffered
  __shared__ _Float16 pS[8][16][136];       // per-wave P: [qrow][key]

  const int q0   = blockIdx.x * 128;
  const int h    = blockIdx.y;
  const int bc   = blockIdx.z;
  const int t    = threadIdx.x;
  const int lane = t & 31;
  const int wv   = t >> 5;
  const int ln15 = lane & 15;
  const int lhi  = lane >> 4;

  const _Float16* Qp = Q + (size_t)(bc * NHH + h) * WW * DHH;
  const _Float16* Kp = K + (size_t)(bc * NHH + h) * WW * DHH;
  const _Float16* Vp = V + ((size_t)bc * FF + (size_t)h * DHH) * WW;

  // ---- Prologue: Q tile + chunk-0 K/V DMAs in flight ----
#if USE_TDM
  if (wv == 0) {
    tdm_load_2d_f16(&qS[0][0],    Qp + (size_t)q0 * DHH, 64, 128, 64, WW, 64, 4, 3);
    tdm_load_2d_f16(&kS[0][0][0], Kp,                    64, 128, 64, WW, 64, 4, 3);
    tdm_load_2d_f16(&vS[0][0][0], Vp,                    128, 64, WW, 64, WW, 5, 3);
  }
#else
#pragma unroll
  for (int i = 0; i < 4; ++i) {
    int u = t + i * 256;                 // 1024 v8h chunks = 128x64
    int row = u >> 3, c8 = (u & 7) << 3;
    *(v8h*)&qS[row][c8] = *(const v8h*)(Qp + (size_t)(q0 + row) * DHH + c8);
  }
#endif

  float m_run[8], l_run[8];
#pragma unroll
  for (int r = 0; r < 8; ++r) { m_run[r] = -INFINITY; l_run[r] = 0.0f; }
  v8f o[4];
#pragma unroll
  for (int j = 0; j < 4; ++j) o[j] = vzero8();

  const int NCHUNK = WW / 128;
  for (int j = 0; j < NCHUNK; ++j) {
    const int buf = j & 1;
    const int k0  = j * 128;
    __syncthreads();  // all waves done reading buf^1 (compute j-1) before DMA reuse
#if USE_TDM
    if (wv == 0) {
      if (j + 1 < NCHUNK) {
        const int kn = k0 + 128;
        tdm_load_2d_f16(&kS[buf ^ 1][0][0], Kp + (size_t)kn * DHH, 64, 128, 64, WW, 64, 4, 3);
        tdm_load_2d_f16(&vS[buf ^ 1][0][0], Vp + kn,               128, 64, WW, 64, WW, 5, 3);
        __builtin_amdgcn_s_wait_tensorcnt(2);  // chunk j (and Q) complete; j+1 in flight
      } else {
        __builtin_amdgcn_s_wait_tensorcnt(0);
      }
    }
#else
#pragma unroll
    for (int i = 0; i < 4; ++i) {
      int u = t + i * 256;
      int row = u >> 3, c8 = (u & 7) << 3;
      *(v8h*)&kS[buf][row][c8] = *(const v8h*)(Kp + (size_t)(k0 + row) * DHH + c8);
    }
#pragma unroll
    for (int i = 0; i < 4; ++i) {
      int u = t + i * 256;                 // 1024 v8h = 64x128
      int dr = u >> 4, c8 = (u & 15) << 3;
      *(v8h*)&vS[buf][dr][c8] = *(const v8h*)(Vp + (size_t)dr * WW + k0 + c8);
    }
#endif
    __syncthreads();  // publish chunk j's tiles

    // S = Q @ K^T for this wave's 16 query rows
    v8f s[8];
#pragma unroll
    for (int jj = 0; jj < 8; ++jj) s[jj] = vzero8();
#pragma unroll
    for (int kk = 0; kk < 2; ++kk) {
      v16h a = load_frag16(&qS[wv * 16 + ln15][kk * 32], lhi);
#pragma unroll
      for (int jj = 0; jj < 8; ++jj) {
        v16h b = load_frag16(&kS[buf][jj * 16 + ln15][kk * 32], lhi);
        s[jj] = wmma_f16(a, b, s[jj]);
      }
    }

    // Online softmax; row m = r + 8*lhi spread over 16 lanes (width-16 reduce)
    float mloc[8];
#pragma unroll
    for (int r = 0; r < 8; ++r) {
      float mv = -INFINITY;
#pragma unroll
      for (int jj = 0; jj < 8; ++jj) mv = fmaxf(mv, s[jj][r]);
#pragma unroll
      for (int off = 8; off > 0; off >>= 1) mv = fmaxf(mv, __shfl_xor(mv, off, 16));
      mloc[r] = mv;
    }
    float scale_o[8], lloc[8];
#pragma unroll
    for (int r = 0; r < 8; ++r) {
      float mnew = fmaxf(m_run[r], mloc[r]);
      scale_o[r] = __expf(m_run[r] - mnew);
      m_run[r] = mnew;
      lloc[r] = 0.0f;
    }
#pragma unroll
    for (int jj = 0; jj < 8; ++jj)
#pragma unroll
      for (int r = 0; r < 8; ++r) {
        float pv = __expf(s[jj][r] - m_run[r]);
        s[jj][r] = pv;
        lloc[r] += pv;
      }
#pragma unroll
    for (int r = 0; r < 8; ++r) {
#pragma unroll
      for (int off = 8; off > 0; off >>= 1) lloc[r] += __shfl_xor(lloc[r], off, 16);
      l_run[r] = l_run[r] * scale_o[r] + lloc[r];
    }
#pragma unroll
    for (int jo = 0; jo < 4; ++jo)
#pragma unroll
      for (int r = 0; r < 8; ++r) o[jo][r] *= scale_o[r];

    // C-layout P -> LDS (A-layout source for P@V); per-wave buffer
#pragma unroll
    for (int jj = 0; jj < 8; ++jj)
#pragma unroll
      for (int r = 0; r < 8; ++r)
        pS[wv][r + 8 * lhi][jj * 16 + ln15] = (_Float16)s[jj][r];
    __syncthreads();

    // O += P @ V
#pragma unroll
    for (int kk = 0; kk < 4; ++kk) {
      v16h a = load_frag16(&pS[wv][ln15][kk * 32], lhi);
#pragma unroll
      for (int jo = 0; jo < 4; ++jo) {
        v16h b = load_frag16(&vS[buf][jo * 16 + ln15][kk * 32], lhi);
        o[jo] = wmma_f16(a, b, o[jo]);
      }
    }
  }

  // Normalize and write back f16 in (f, w) layout for the output projection.
  _Float16* Ap = A + (size_t)bc * FF * WW;
#pragma unroll
  for (int jo = 0; jo < 4; ++jo) {
    int d = jo * 16 + ln15;
#pragma unroll
    for (int r = 0; r < 8; ++r) {
      int qrow = q0 + wv * 16 + 8 * lhi + r;
      Ap[(size_t)(h * DHH + d) * WW + qrow] = (_Float16)(o[jo][r] / l_run[r]);
    }
  }
}

// ---------------------------------------------------------------------------
extern "C" void kernel_launch(void* const* d_in, const int* in_sizes, int n_in,
                              void* d_out, int out_size, void* d_ws, size_t ws_size,
                              hipStream_t stream) {
  const float* x  = (const float*)d_in[0];
  const float* wq = (const float*)d_in[1];
  const float* bq = (const float*)d_in[2];
  const float* wk = (const float*)d_in[3];
  const float* bk = (const float*)d_in[4];
  const float* wv = (const float*)d_in[5];
  const float* bv = (const float*)d_in[6];
  const float* wo = (const float*)d_in[7];
  const float* bo = (const float*)d_in[8];
  const float* fr = (const float*)d_in[9];
  float* out = (float*)d_out;

  const size_t planesz = (size_t)BCC * FF * WW;  // 4 Mi elements (f16 -> 8 MB)
  _Float16* qw = (_Float16*)d_ws;
  _Float16* kw = qw + planesz;
  _Float16* vw = kw + planesz;
  _Float16* aw = vw + planesz;

  dim3 gg(WW / 128, FF / 128, BCC);
  mc_gemm_wmma<true,  true,  float,    _Float16><<<gg, 256, 0, stream>>>(wq, bq, x, qw);
  mc_gemm_wmma<true,  false, float,    _Float16><<<gg, 256, 0, stream>>>(wk, bk, x, kw);
  mc_gemm_wmma<false, false, float,    _Float16><<<gg, 256, 0, stream>>>(wv, bv, x, vw);

  int rope_threads = BCC * NHH * WW * (DHH / 2);
  rope_kernel<<<rope_threads / 256, 256, 0, stream>>>(qw, kw, fr);

  attn_wmma<<<dim3(WW / 128, NHH, BCC), 256, 0, stream>>>(qw, kw, vw, aw);

  mc_gemm_wmma<false, false, _Float16, float   ><<<gg, 256, 0, stream>>>(wo, bo, aw, out);
}
